// M2G4RTP_38809324487447
// MI455X (gfx1250) — matrix operand order repaired and checked by reading, not compiled, not run on hardware
//
#include <hip/hip_runtime.h>
#include <math.h>

// Problem dims (match reference)
#define B_   512
#define S_   100
#define D_   512
#define SF_  5
#define ND4  (4 * D_)

typedef __attribute__((ext_vector_type(16))) _Float16 v16h;
typedef __attribute__((ext_vector_type(8)))  float    v8f;
typedef __attribute__((ext_vector_type(2)))  _Float16 h2;
typedef __attribute__((ext_vector_type(4)))  _Float16 h4;

// ---- WMMA GEMM tile config: block = 8 waves = 256 threads.
// Block tile 64(M) x 64(N), K-step 32. Each wave owns a 16x32 output slab:
// one A fragment feeds two WMMAs per K-step.
#define BM 64
#define BN 64
#define BK 32
#define SPAD 8

__device__ __forceinline__ float sigm(float x) { return 1.0f / (1.0f + expf(-x)); }

__device__ __forceinline__ float wred_sum(float v) {
#pragma unroll
    for (int off = 16; off; off >>= 1) v += __shfl_xor(v, off, 32);
    return v;
}
__device__ __forceinline__ float wred_max(float v) {
#pragma unroll
    for (int off = 16; off; off >>= 1) v = fmaxf(v, __shfl_xor(v, off, 32));
    return v;
}

// ---------------------------------------------------------------------------
// Fused GEMM: C[m,n] = sum_k A1[m,k]*W1[n,k] + b1[n]
//             (+ sum_k A2[m,k]*W2[n,k] + b2[n]   when DUAL)
// A row-major [M,K], W row-major [N,K] (torch weight layout), f32 in memory,
// f16 WMMA inputs, f32 accumulate.  OUTH selects f16 output (for the
// L2-resident context projections) vs f32.
// Staging: float4 global loads -> h4 packed 8-byte LDS stores.
// Per-lane fragment layouts follow CDNA5 ISA 7.12.2 (wave32).
// ---------------------------------------------------------------------------
template <int KK, bool DUAL, bool OUTH>
__global__ __launch_bounds__(256)
void wmma_gemm(const float* __restrict__ A1, const float* __restrict__ W1,
               const float* __restrict__ A2, const float* __restrict__ W2,
               const float* __restrict__ b1, const float* __restrict__ b2,
               void* __restrict__ Cout, int M, int N)
{
    __shared__ _Float16 sA[BM][BK + SPAD];
    __shared__ _Float16 sW[BN][BK + SPAD];

    const int tid  = threadIdx.x;
    const int lane = tid & 31;
    const int wave = tid >> 5;
    const int wm   = wave & 3;           // 4 wave-tiles along M (16 each)
    const int nb   = (wave >> 2) * 32;   // 2 wave-slabs along N (32 each)
    const int m0   = blockIdx.y * BM;
    const int n0   = blockIdx.x * BN;

    const int row = lane & 15;   // M-row (A) / N-col (B) within 16
    const int hb  = lane >> 4;   // lane half

    v8f acc0 = {};
    v8f acc1 = {};

    constexpr int NPAIR = DUAL ? 2 : 1;
    const float* As[2] = {A1, A2};
    const float* Ws[2] = {W1, W2};

#pragma unroll 1
    for (int pair = 0; pair < NPAIR; ++pair) {
        const float* __restrict__ A = As[pair];
        const float* __restrict__ W = Ws[pair];
#pragma unroll 2
        for (int k0 = 0; k0 < KK; k0 += BK) {
            // Stage A tile (64x32 f32 -> f16): float4 load + packed h4 store
#pragma unroll
            for (int e = 0; e < (BM * BK) / (256 * 4); ++e) {
                int p = tid + e * 256;     // quad index (8 quads per row)
                int i = p >> 3, j = (p & 7) * 4;
                const float4 v = *(const float4*)&A[(size_t)(m0 + i) * KK + k0 + j];
                h4 pk;
                pk[0] = (_Float16)v.x; pk[1] = (_Float16)v.y;
                pk[2] = (_Float16)v.z; pk[3] = (_Float16)v.w;
                *(h4*)&sA[i][j] = pk;
            }
            // Stage W tile (64x32)
#pragma unroll
            for (int e = 0; e < (BN * BK) / (256 * 4); ++e) {
                int p = tid + e * 256;
                int i = p >> 3, j = (p & 7) * 4;
                const float4 v = *(const float4*)&W[(size_t)(n0 + i) * KK + k0 + j];
                h4 pk;
                pk[0] = (_Float16)v.x; pk[1] = (_Float16)v.y;
                pk[2] = (_Float16)v.z; pk[3] = (_Float16)v.w;
                *(h4*)&sW[i][j] = pk;
            }
            __syncthreads();

            v16h af, bf0, bf1;
#pragma unroll
            for (int idx = 0; idx < 16; ++idx) {
                // A 16x32 f16 layout: K = (idx>>3)*16 + halfwave*8 + (idx&7)
                int ka = ((idx >> 3) << 4) + (hb << 3) + (idx & 7);
                af[idx] = sA[wm * 16 + row][ka];
                // B 32x16 f16 layout: K = (idx>>3)*16 + halfwave*8 + (idx&7), N = lane&15
                int kb = ((idx >> 3) << 4) + (hb << 3) + (idx & 7);
                bf0[idx] = sW[nb + row][kb];
                bf1[idx] = sW[nb + 16 + row][kb];
            }
            acc0 = __builtin_amdgcn_wmma_f32_16x16x32_f16(
                false, af, false, bf0, (short)0, acc0, false, false);
            acc1 = __builtin_amdgcn_wmma_f32_16x16x32_f16(
                false, af, false, bf1, (short)0, acc1, false, false);
            __syncthreads();
        }
    }

    // Epilogue: C/D layout -> VGPR r holds (M = r + 8*halfwave, N = lane&15)
    const int mBase = m0 + wm * 16 + (hb << 3);
    const int nA = n0 + nb + (lane & 15);
    const int nB = nA + 16;
    float biasA = b1[nA], biasB = b1[nB];
    if (DUAL) { biasA += b2[nA]; biasB += b2[nB]; }

    if (OUTH) {
        _Float16* C = (_Float16*)Cout;
#pragma unroll
        for (int r = 0; r < 8; ++r) {
            size_t m = mBase + r;
            C[m * N + nA] = (_Float16)(acc0[r] + biasA);
            C[m * N + nB] = (_Float16)(acc1[r] + biasB);
        }
    } else {
        float* C = (float*)Cout;
#pragma unroll
        for (int r = 0; r < 8; ++r) {
            size_t m = mBase + r;
            C[m * N + nA] = acc0[r] + biasA;
            C[m * N + nB] = acc1[r] + biasB;
        }
    }
}

// ---------------------------------------------------------------------------
// fn_emb = start_fea @ fne_W.T   (K = 5, too small for WMMA)
// ---------------------------------------------------------------------------
__global__ __launch_bounds__(256)
void fnemb_kernel(const float* __restrict__ sf, const float* __restrict__ Wf,
                  float* __restrict__ out)
{
    int i = blockIdx.x * 256 + threadIdx.x;   // over B*D
    int b = i / D_, d = i % D_;
    float acc = 0.0f;
#pragma unroll
    for (int k = 0; k < SF_; ++k) acc += sf[b * SF_ + k] * Wf[d * SF_ + k];
    out[i] = acc;
}

// ---------------------------------------------------------------------------
// LSTM pointwise: gates [B,4D] (i,f,g,o), in-place-safe c update
// ---------------------------------------------------------------------------
__global__ __launch_bounds__(256)
void lstm_update(const float* __restrict__ gates, const float* __restrict__ c_in,
                 float* __restrict__ h_out, float* __restrict__ c_out)
{
    int i = blockIdx.x * 256 + threadIdx.x;   // over B*D
    int b = i / D_, d = i % D_;
    const float* g = gates + (size_t)b * ND4;
    float ig = g[d], fg = g[D_ + d], gg = g[2 * D_ + d], og = g[3 * D_ + d];
    float c2 = sigm(fg) * c_in[i] + sigm(ig) * tanhf(gg);
    c_out[i] = c2;
    h_out[i] = sigm(og) * tanhf(c2);
}

// ---------------------------------------------------------------------------
// Per-call state init: copy inputs into mutable ws buffers, build mask0
// (with _mask_modify), zero cur_eta.  One block per batch row.
// ---------------------------------------------------------------------------
__global__ __launch_bounds__(256)
void init_state(const float* __restrict__ dec0, const float* __restrict__ h0,
                const float* __restrict__ c0, const int* __restrict__ vmask,
                float* __restrict__ dec_in, float* __restrict__ h_buf,
                float* __restrict__ c_buf, float* __restrict__ ce_buf,
                float* __restrict__ cur_eta, unsigned char* __restrict__ mask)
{
    int b = blockIdx.x;
    for (int d = threadIdx.x; d < D_; d += 256) {
        size_t i = (size_t)b * D_ + d;
        dec_in[i] = dec0[i];
        h_buf[i]  = h0[i];
        c_buf[i]  = c0[i];
        ce_buf[i] = c0[i];
    }
    __shared__ int notall;
    if (threadIdx.x == 0) { notall = 0; cur_eta[b] = 0.0f; }
    __syncthreads();
    if (threadIdx.x < S_) {
        unsigned char m = (vmask[b * S_ + threadIdx.x] != 0) ? 1 : 0;
        mask[b * S_ + threadIdx.x] = m;
        if (!m) atomicOr(&notall, 1);
    }
    __syncthreads();
    if (threadIdx.x == 0 && notall == 0) mask[b * S_ + (S_ - 1)] = 0;
}

// ---------------------------------------------------------------------------
// Glimpse attention. One block per batch row.
//   u[s]  = sum_d v[d]*tanh(q[b,d] + e[s,b,d]);  masked -> -inf
//   p     = softmax(u);  g_out[b,d] = sum_s p[s]*e[s,b,d]
// e is f16 (L2-resident), read as f16x2 pairs; q/v cached in LDS.
// ---------------------------------------------------------------------------
__global__ __launch_bounds__(256)
void glimpse_kernel(const float* __restrict__ q, const _Float16* __restrict__ e,
                    const float* __restrict__ vv, const unsigned char* __restrict__ mask,
                    float* __restrict__ gout)
{
    int b = blockIdx.x;
    __shared__ float sq[D_];
    __shared__ float sv[D_];
    __shared__ float su[S_];
    for (int d = threadIdx.x; d < D_; d += 256) { sq[d] = q[(size_t)b * D_ + d]; sv[d] = vv[d]; }
    __syncthreads();

    int lane = threadIdx.x & 31, wave = threadIdx.x >> 5;
    for (int s = wave; s < S_; s += 8) {
        const h2* er2 = (const h2*)(e + ((size_t)s * B_ + b) * D_);
        float acc = 0.0f;
        for (int p = lane; p < D_ / 2; p += 32) {
            h2 ev = er2[p];
            acc += sv[2 * p]     * tanhf(sq[2 * p]     + (float)ev[0])
                 + sv[2 * p + 1] * tanhf(sq[2 * p + 1] + (float)ev[1]);
        }
        acc = wred_sum(acc);
        if (lane == 0) su[s] = mask[b * S_ + s] ? -__builtin_inff() : acc;
    }
    __syncthreads();

    if (wave == 0) {  // softmax over 100 entries, wave-synchronous
        float mx = -__builtin_inff();
        for (int s = lane; s < S_; s += 32) mx = fmaxf(mx, su[s]);
        mx = wred_max(mx);
        float sum = 0.0f;
        for (int s = lane; s < S_; s += 32) { float ex = expf(su[s] - mx); su[s] = ex; sum += ex; }
        sum = wred_sum(sum);
        float inv = 1.0f / sum;
        for (int s = lane; s < S_; s += 32) su[s] *= inv;
    }
    __syncthreads();

    {   // g_l: each thread owns adjacent d pair (f16x2 loads)
        int d0 = threadIdx.x * 2;
        float a0 = 0.0f, a1 = 0.0f;
        for (int s = 0; s < S_; ++s) {
            h2 ev = *(const h2*)(e + ((size_t)s * B_ + b) * D_ + d0);
            float p = su[s];
            a0 += p * (float)ev[0];
            a1 += p * (float)ev[1];
        }
        gout[(size_t)b * D_ + d0]     = a0;
        gout[(size_t)b * D_ + d0 + 1] = a1;
    }
}

// ---------------------------------------------------------------------------
// Pointer attention + log_softmax + greedy argmax. One block per batch row.
//   logits[s] = mask ? -inf : C_CLIP * tanh(u_p[s])
//   log_p written to out[b, t, :]; argmax -> idxs[b]; sels out as float.
// ---------------------------------------------------------------------------
__global__ __launch_bounds__(256)
void pointer_kernel(const float* __restrict__ q, const _Float16* __restrict__ e,
                    const float* __restrict__ vv, const unsigned char* __restrict__ mask,
                    float* __restrict__ logp_out, float* __restrict__ sels_out,
                    int* __restrict__ idxs, int t)
{
    int b = blockIdx.x;
    __shared__ float sq[D_];
    __shared__ float sv[D_];
    __shared__ float su[S_];
    for (int d = threadIdx.x; d < D_; d += 256) { sq[d] = q[(size_t)b * D_ + d]; sv[d] = vv[d]; }
    __syncthreads();

    int lane = threadIdx.x & 31, wave = threadIdx.x >> 5;
    for (int s = wave; s < S_; s += 8) {
        const h2* er2 = (const h2*)(e + ((size_t)s * B_ + b) * D_);
        float acc = 0.0f;
        for (int p = lane; p < D_ / 2; p += 32) {
            h2 ev = er2[p];
            acc += sv[2 * p]     * tanhf(sq[2 * p]     + (float)ev[0])
                 + sv[2 * p + 1] * tanhf(sq[2 * p + 1] + (float)ev[1]);
        }
        acc = wred_sum(acc);
        if (lane == 0) su[s] = mask[b * S_ + s] ? -__builtin_inff() : 10.0f * tanhf(acc);
    }
    __syncthreads();

    if (wave == 0) {
        float mx = -__builtin_inff();
        for (int s = lane; s < S_; s += 32) mx = fmaxf(mx, su[s]);
        mx = wred_max(mx);
        float sum = 0.0f;
        for (int s = lane; s < S_; s += 32) sum += expf(su[s] - mx);
        sum = wred_sum(sum);
        float lse = mx + logf(sum);

        float bv = -__builtin_inff();
        int   bi = S_;
        for (int s = lane; s < S_; s += 32) {
            float lg = su[s];
            logp_out[((size_t)b * S_ + t) * S_ + s] = lg - lse;
            if (lg > bv) { bv = lg; bi = s; }   // first-max within lane (s increasing)
        }
        // cross-lane argmax, ties -> smaller index (matches jnp.argmax)
#pragma unroll
        for (int off = 16; off; off >>= 1) {
            float ov = __shfl_xor(bv, off, 32);
            int   oi = __shfl_xor(bi, off, 32);
            if (ov > bv || (ov == bv && oi < bi)) { bv = ov; bi = oi; }
        }
        if (lane == 0) {
            idxs[b] = bi;
            sels_out[(size_t)b * S_ + t] = (float)bi;
        }
    }
}

// ---------------------------------------------------------------------------
// Eta/gather: cur_emb = context[idx,b,:], dec_next = embedded[idx,b,:],
// eta = [last_node, cur_emb] @ Wl^T + bl; cur_eta += eta; mask update with
// _mask_modify.  last_node <- cur_emb (and doubles as eta-LSTM x input).
// One block per batch row.
// ---------------------------------------------------------------------------
__global__ __launch_bounds__(256)
void eta_gather(const int* __restrict__ idxs, const float* __restrict__ context,
                const float* __restrict__ emb, const float* __restrict__ Wl,
                const float* __restrict__ bl, float* __restrict__ last_node,
                float* __restrict__ dec_in, float* __restrict__ cur_eta,
                unsigned char* __restrict__ mask, float* __restrict__ etas_out, int t)
{
    int b = blockIdx.x;
    int idx = idxs[b];
    const float* ctx = context + ((size_t)idx * B_ + b) * D_;
    const float* em  = emb     + ((size_t)idx * B_ + b) * D_;

    __shared__ float part[8];
    __shared__ int notall;
    if (threadIdx.x == 0) notall = 0;

    float acc = 0.0f;
    for (int d = threadIdx.x; d < D_; d += 256) {
        float ln = last_node[(size_t)b * D_ + d];
        float ce = ctx[d];
        acc += ln * Wl[d] + ce * Wl[D_ + d];
        last_node[(size_t)b * D_ + d] = ce;     // becomes eta-LSTM input x
        dec_in[(size_t)b * D_ + d]    = em[d];  // next main-LSTM input
    }
    acc = wred_sum(acc);
    int lane = threadIdx.x & 31, wave = threadIdx.x >> 5;
    if (lane == 0) part[wave] = acc;
    __syncthreads();
    if (threadIdx.x == 0) {
        float s = bl[0];
#pragma unroll
        for (int w = 0; w < 8; ++w) s += part[w];
        float v = cur_eta[b] + s;
        cur_eta[b] = v;
        etas_out[(size_t)b * S_ + t] = v;
        mask[b * S_ + idx] = 1;                 // update_mask
    }
    __syncthreads();
    if (threadIdx.x < S_) {
        if (!mask[b * S_ + threadIdx.x]) atomicOr(&notall, 1);
    }
    __syncthreads();
    if (threadIdx.x == 0 && notall == 0) mask[b * S_ + (S_ - 1)] = 0;  // _mask_modify
}

// ---------------------------------------------------------------------------
// Host-side orchestration
// ---------------------------------------------------------------------------
extern "C" void kernel_launch(void* const* d_in, const int* in_sizes, int n_in,
                              void* d_out, int out_size, void* d_ws, size_t ws_size,
                              hipStream_t stream)
{
    (void)in_sizes; (void)n_in; (void)out_size; (void)ws_size;

    const float* start_fea   = (const float*)d_in[0];
    const float* decoder_in  = (const float*)d_in[1];
    const float* embedded    = (const float*)d_in[2];   // [S,B,D]
    const float* h0          = (const float*)d_in[3];
    const float* c0          = (const float*)d_in[4];
    const float* context     = (const float*)d_in[5];   // [S,B,D]
    const int*   vmask       = (const int*)  d_in[6];   // [B,S]
    // d_in[7] = V : unused by the reference outputs
    const float* lstm_Wih = (const float*)d_in[8];
    const float* lstm_Whh = (const float*)d_in[9];
    const float* lstm_bih = (const float*)d_in[10];
    const float* lstm_bhh = (const float*)d_in[11];
    const float* eta_Wih  = (const float*)d_in[12];
    const float* eta_Whh  = (const float*)d_in[13];
    const float* eta_bih  = (const float*)d_in[14];
    const float* eta_bhh  = (const float*)d_in[15];
    const float* ptr_Wq   = (const float*)d_in[16];
    const float* ptr_bq   = (const float*)d_in[17];
    const float* ptr_Wr   = (const float*)d_in[18];
    const float* ptr_br   = (const float*)d_in[19];
    const float* ptr_v    = (const float*)d_in[20];
    const float* gl_Wq    = (const float*)d_in[21];
    const float* gl_bq    = (const float*)d_in[22];
    const float* gl_Wr    = (const float*)d_in[23];
    const float* gl_br    = (const float*)d_in[24];
    const float* gl_v     = (const float*)d_in[25];
    const float* fne_W    = (const float*)d_in[26];
    const float* eta_linW = (const float*)d_in[27];
    const float* eta_linb = (const float*)d_in[28];

    // ---- workspace carve-up (256B aligned) ----
    char* wsp = (char*)d_ws;
    auto carve = [&](size_t bytes) -> char* {
        char* p = wsp;
        wsp += (bytes + 255) & ~(size_t)255;
        return p;
    };
    const size_t SBD = (size_t)S_ * B_ * D_;
    _Float16* e_g   = (_Float16*)carve(SBD * sizeof(_Float16));  // 52.4 MB (L2-resident)
    _Float16* e_p   = (_Float16*)carve(SBD * sizeof(_Float16));  // 52.4 MB
    float* gates    = (float*)carve((size_t)B_ * ND4 * sizeof(float));
    float* h_buf    = (float*)carve((size_t)B_ * D_ * sizeof(float));
    float* c_buf    = (float*)carve((size_t)B_ * D_ * sizeof(float));
    float* he_buf   = (float*)carve((size_t)B_ * D_ * sizeof(float));
    float* ce_buf   = (float*)carve((size_t)B_ * D_ * sizeof(float));
    float* dec_in   = (float*)carve((size_t)B_ * D_ * sizeof(float));
    float* last_nd  = (float*)carve((size_t)B_ * D_ * sizeof(float));
    float* g_l      = (float*)carve((size_t)B_ * D_ * sizeof(float));
    float* q_buf    = (float*)carve((size_t)B_ * D_ * sizeof(float));
    float* cur_eta  = (float*)carve((size_t)B_ * sizeof(float));
    int*   idxs     = (int*)  carve((size_t)B_ * sizeof(int));
    unsigned char* mask = (unsigned char*)carve((size_t)B_ * S_);

    float* out      = (float*)d_out;
    float* out_logp = out;                                 // [B, S_, S_]
    float* out_sels = out + (size_t)B_ * S_ * S_;          // [B, S_]
    float* out_etas = out_sels + (size_t)B_ * S_;          // [B, S_]

    const dim3 blk(256);
    const int  bdGrid = (B_ * D_) / 256;                   // 1024 blocks

    // 1) state init + mask0 (with _mask_modify)
    init_state<<<B_, blk, 0, stream>>>(decoder_in, h0, c0, vmask,
                                       dec_in, h_buf, c_buf, ce_buf, cur_eta, mask);

    // 2) fn_emb = start_fea @ fne_W.T  -> last_node (also eta warm-up x)
    fnemb_kernel<<<bdGrid, blk, 0, stream>>>(start_fea, fne_W, last_nd);

    // 3) precompute context projections, stored f16 so both stay L2-resident
    {
        dim3 grid(D_ / BN, (S_ * B_) / BM);  // 8 x 800
        wmma_gemm<D_, false, true><<<grid, blk, 0, stream>>>(
            context, gl_Wr, nullptr, nullptr, gl_br, nullptr, e_g, S_ * B_, D_);
        wmma_gemm<D_, false, true><<<grid, blk, 0, stream>>>(
            context, ptr_Wr, nullptr, nullptr, ptr_br, nullptr, e_p, S_ * B_, D_);
    }

    // 4) eta-LSTM warm-up: (he,ce) = lstm(fn_emb, h0, c0)
    dim3 gridGates(ND4 / BN, B_ / BM);  // 32 x 8
    dim3 gridProj(D_ / BN, B_ / BM);    // 8 x 8
    wmma_gemm<D_, true, false><<<gridGates, blk, 0, stream>>>(
        last_nd, eta_Wih, h_buf, eta_Whh, eta_bih, eta_bhh, gates, B_, ND4);
    lstm_update<<<bdGrid, blk, 0, stream>>>(gates, ce_buf, he_buf, ce_buf);

    // 5) sequential decode: 100 steps
    for (int t = 0; t < S_; ++t) {
        // main LSTM: gates = dec_in@Wih^T + h@Whh^T + bih + bhh
        wmma_gemm<D_, true, false><<<gridGates, blk, 0, stream>>>(
            dec_in, lstm_Wih, h_buf, lstm_Whh, lstm_bih, lstm_bhh, gates, B_, ND4);
        lstm_update<<<bdGrid, blk, 0, stream>>>(gates, c_buf, h_buf, c_buf);

        // glimpse: q_g = hy@gl_Wq^T + gl_bq ; attention -> g_l
        wmma_gemm<D_, false, false><<<gridProj, blk, 0, stream>>>(
            h_buf, gl_Wq, nullptr, nullptr, gl_bq, nullptr, q_buf, B_, D_);
        glimpse_kernel<<<B_, blk, 0, stream>>>(q_buf, e_g, gl_v, mask, g_l);

        // pointer: q_p = g_l@ptr_Wq^T + ptr_bq ; logits/log_softmax/argmax
        wmma_gemm<D_, false, false><<<gridProj, blk, 0, stream>>>(
            g_l, ptr_Wq, nullptr, nullptr, ptr_bq, nullptr, q_buf, B_, D_);
        pointer_kernel<<<B_, blk, 0, stream>>>(q_buf, e_p, ptr_v, mask,
                                               out_logp, out_sels, idxs, t);

        // gather cur_emb/dec_next, eta linear, mask update
        eta_gather<<<B_, blk, 0, stream>>>(idxs, context, embedded, eta_linW, eta_linb,
                                           last_nd, dec_in, cur_eta, mask, out_etas, t);

        // eta LSTM: gates = cur_emb@eta_Wih^T + he@eta_Whh^T + biases
        wmma_gemm<D_, true, false><<<gridGates, blk, 0, stream>>>(
            last_nd, eta_Wih, he_buf, eta_Whh, eta_bih, eta_bhh, gates, B_, ND4);
        lstm_update<<<bdGrid, blk, 0, stream>>>(gates, ce_buf, he_buf, ce_buf);
    }
}
